// MSDeformableAttention_12180527251512
// MI455X (gfx1250) — matrix operand order repaired and checked by reading, not compiled, not run on hardware
//
#include <hip/hip_runtime.h>
#include <hip/hip_bf16.h>

typedef __attribute__((ext_vector_type(16))) _Float16 v16h;
typedef __attribute__((ext_vector_type(8)))  _Float16 v8h;
typedef __attribute__((ext_vector_type(4)))  _Float16 v4h;
typedef __attribute__((ext_vector_type(8)))  float    v8f;
typedef __attribute__((ext_vector_type(4)))  float    f4;

#define EMBED_DIM 256
#define NUM_HEADS 8
#define NUM_LEVELS 4
#define NUM_POINTS 4
#define HEAD_DIM 32
#define BS 8
#define LEN_Q 1000
#define LEN_V 13294

// ---------------- WMMA GEMM ----------------
// C[M,N] = A[M,K](f32) * B[K,N](f32) + bias[N]
// MODE 0: row-major C.   MODE 1: value projection stored transposed to
//         [b][head][spatial][HEAD_DIM] (spatial length == LEN_V).
#define BM 128
#define BN 64
#define BK 32

template <int MODE>
__global__ __launch_bounds__(256) void gemm_f16_wmma(
    const float* __restrict__ A, const float* __restrict__ B,
    const float* __restrict__ bias, float* __restrict__ C,
    int M, int N, int K)
{
    // A tile row-major; B tile stored TRANSPOSED so both fragments are
    // contiguous ds_load_b128 pairs.
    __shared__ alignas(16) _Float16 As[BM][BK + 8];
    __shared__ alignas(16) _Float16 BsT[BN][BK + 8];

    const int tid   = threadIdx.x;
    const int lane  = tid & 31;
    const int wave  = tid >> 5;     // 0..7
    const int waveM = wave >> 1;    // 0..3  (rows of 32)
    const int waveN = wave & 1;     // 0..1  (cols of 32)
    const int blockM = blockIdx.x * BM;
    const int blockN = blockIdx.y * BN;

    const int grp = lane >> 4;      // half-wave group (ISA 16-bit layouts)
    const int l16 = lane & 15;

    // ---- staging assignments (all loads unconditional; A rows clamped) ----
    const int aCol  = (tid & 7) * 4;    // 0..28, float4 column in k-tile
    const int aRow0 = tid >> 3;         // 0..31, +32*i
    const int bCol  = (tid & 15) * 4;   // 0..60, float4 column in n-tile
    const int bRow0 = tid >> 4;         // 0..15, +16*i

    // Hoisted, clamped base pointers: only bumped by BK / BK*N per k-step,
    // so the hot loop has no 64-bit address multiplies.
    const float* aPtr[4];
    #pragma unroll
    for (int i = 0; i < 4; ++i) {
        int gm = blockM + aRow0 + 32 * i;
        gm = (gm < M) ? gm : (M - 1);              // clamp, never predicate
        aPtr[i] = A + (long long)gm * K + aCol;
    }
    const float* bPtr[2];
    #pragma unroll
    for (int i = 0; i < 2; ++i)
        bPtr[i] = B + (long long)(bRow0 + 16 * i) * N + blockN + bCol;

    f4 aReg[4];
    f4 bReg[2];

    auto load_regs = [&]() {
        #pragma unroll
        for (int i = 0; i < 4; ++i) {
            aReg[i] = *(const f4*)aPtr[i];
            aPtr[i] += BK;
        }
        #pragma unroll
        for (int i = 0; i < 2; ++i) {
            bReg[i] = *(const f4*)bPtr[i];
            bPtr[i] += (long long)BK * N;
        }
    };

    auto store_lds = [&]() {
        #pragma unroll
        for (int i = 0; i < 4; ++i) {
            v4h h = { (_Float16)aReg[i].x, (_Float16)aReg[i].y,
                      (_Float16)aReg[i].z, (_Float16)aReg[i].w };
            *(v4h*)&As[aRow0 + 32 * i][aCol] = h;
        }
        #pragma unroll
        for (int i = 0; i < 2; ++i) {
            int r = bRow0 + 16 * i;
            BsT[bCol + 0][r] = (_Float16)bReg[i].x;
            BsT[bCol + 1][r] = (_Float16)bReg[i].y;
            BsT[bCol + 2][r] = (_Float16)bReg[i].z;
            BsT[bCol + 3][r] = (_Float16)bReg[i].w;
        }
    };

    v8f acc[2][2] = {};

    load_regs();

    for (int k0 = 0; k0 < K; k0 += BK) {
        store_lds();
        __syncthreads();

        if (k0 + BK < K) {
            load_regs();                           // overlap with WMMA below
            if (k0 + 2 * BK < K)                   // GL2 prefetch 1 tile ahead
                __builtin_prefetch(aPtr[0], 0, 1);
        }

        // ---- fragments: 2x ds_load_b128 each ----
        v16h af[2], bf[2];
        #pragma unroll
        for (int mi = 0; mi < 2; ++mi) {
            int mrow = waveM * 32 + mi * 16 + l16;
            v8h lo = *(const v8h*)&As[mrow][grp * 8];        // K = grp*8 .. +7
            v8h hi = *(const v8h*)&As[mrow][16 + grp * 8];   // K = 16+grp*8 .. +7
            af[mi] = __builtin_shufflevector(lo, hi, 0,1,2,3,4,5,6,7,
                                                     8,9,10,11,12,13,14,15);
        }
        #pragma unroll
        for (int ni = 0; ni < 2; ++ni) {
            int ncol = waveN * 32 + ni * 16 + l16;
            v8h lo = *(const v8h*)&BsT[ncol][grp * 8];       // K = grp*8 .. +7
            v8h hi = *(const v8h*)&BsT[ncol][16 + grp * 8];  // K = 16+grp*8 .. +7
            bf[ni] = __builtin_shufflevector(lo, hi, 0,1,2,3,4,5,6,7,
                                                     8,9,10,11,12,13,14,15);
        }

        #pragma unroll
        for (int mi = 0; mi < 2; ++mi)
            #pragma unroll
            for (int ni = 0; ni < 2; ++ni)
                acc[mi][ni] = __builtin_amdgcn_wmma_f32_16x16x32_f16(
                    false, af[mi], false, bf[ni],
                    (short)0, acc[mi][ni], false, false);
        __syncthreads();
    }

    // ---- epilogue: C/D layout -> lane l16 = N, vgpr r -> M = r + grp*8 ----
    #pragma unroll
    for (int mi = 0; mi < 2; ++mi) {
        #pragma unroll
        for (int ni = 0; ni < 2; ++ni) {
            int n = blockN + waveN * 32 + ni * 16 + l16;
            float bv = bias[n];
            #pragma unroll
            for (int r = 0; r < 8; ++r) {
                int m = blockM + waveM * 32 + mi * 16 + grp * 8 + r;
                if (m < M) {
                    float v = acc[mi][ni][r] + bv;
                    if (MODE == 0) {
                        C[(long long)m * N + n] = v;
                    } else {
                        int b = m / LEN_V, s = m % LEN_V;   // const divide
                        int h = n >> 5, d = n & 31;
                        C[(((long long)(b * NUM_HEADS + h)) * LEN_V + s)
                              * HEAD_DIM + d] = v;
                    }
                }
            }
        }
    }
}

// ---------------- softmax + sampling-location prep ----------------
__global__ __launch_bounds__(256) void prep_kernel(
    const float* __restrict__ off_raw,   // [BS*LEN_Q, 256]
    const float* __restrict__ attn_raw,  // [BS*LEN_Q, 128]
    const float* __restrict__ ref,       // [BS, LEN_Q, 4, 2]
    float* __restrict__ loc,             // [BS*LEN_Q*8, 4, 4, 2]
    float* __restrict__ attnw)           // [BS*LEN_Q*8, 4, 4]
{
    int idx = blockIdx.x * blockDim.x + threadIdx.x;  // (b*LEN_Q+q)*8 + h
    if (idx >= BS * LEN_Q * NUM_HEADS) return;
    int h  = idx & 7;
    int bq = idx >> 3;

    const float Wn[4] = {100.f, 50.f, 25.f, 13.f};
    const float Hn[4] = {100.f, 50.f, 25.f, 13.f};

    float logits[16];
    float mx = -3.0e38f;
    #pragma unroll
    for (int i = 0; i < 16; ++i) {
        logits[i] = attn_raw[(long long)bq * 128 + h * 16 + i];
        mx = fmaxf(mx, logits[i]);
    }
    float sum = 0.f;
    #pragma unroll
    for (int i = 0; i < 16; ++i) {
        logits[i] = __expf(logits[i] - mx);
        sum += logits[i];
    }
    float inv = 1.0f / sum;

    #pragma unroll
    for (int l = 0; l < NUM_LEVELS; ++l) {
        float rx = ref[((long long)bq * NUM_LEVELS + l) * 2 + 0];
        float ry = ref[((long long)bq * NUM_LEVELS + l) * 2 + 1];
        #pragma unroll
        for (int p = 0; p < NUM_POINTS; ++p) {
            int col = (((h * NUM_LEVELS + l) * NUM_POINTS) + p) * 2;
            float ox = off_raw[(long long)bq * 256 + col + 0];
            float oy = off_raw[(long long)bq * 256 + col + 1];
            long long li = ((long long)idx * NUM_LEVELS + l) * NUM_POINTS + p;
            loc[li * 2 + 0]  = rx + ox / Wn[l];
            loc[li * 2 + 1]  = ry + oy / Hn[l];
            attnw[li]        = logits[l * NUM_POINTS + p] * inv;
        }
    }
}

// ---------------- bilinear gather + attention-weighted sum ----------------
__device__ __forceinline__ float corner_fetch(const float* __restrict__ vl,
                                              int y, int x, int H, int W, int lane)
{
    if (y < 0 || y >= H || x < 0 || x >= W) return 0.0f;
    return vl[((long long)y * W + x) * HEAD_DIM + lane];
}

__global__ __launch_bounds__(256) void sample_kernel(
    const float* __restrict__ vproj,   // [BS][8][LEN_V][32]
    const float* __restrict__ loc,
    const float* __restrict__ attnw,
    float* __restrict__ mid)           // [BS*LEN_Q, 256]
{
    const int lane = threadIdx.x & 31;         // == head channel d
    const int wave = threadIdx.x >> 5;
    const int gid  = blockIdx.x * 8 + wave;    // (b*LEN_Q+q)*8 + h
    if (gid >= BS * LEN_Q * NUM_HEADS) return;
    const int h  = gid & 7;
    const int bq = gid >> 3;
    const int b  = bq / LEN_Q;

    const int Hi[4] = {100, 50, 25, 13};
    const int Wi[4] = {100, 50, 25, 13};
    const int st[4] = {0, 10000, 12500, 13125};

    const float* __restrict__ vb =
        vproj + (long long)(b * NUM_HEADS + h) * LEN_V * HEAD_DIM;

    float acc = 0.0f;
    #pragma unroll
    for (int l = 0; l < NUM_LEVELS; ++l) {
        const int H = Hi[l], W = Wi[l];
        const float* __restrict__ vl = vb + (long long)st[l] * HEAD_DIM;

        // vectorized, 16B/32B-aligned broadcast loads for this level
        const f4* locv = (const f4*)(loc + (long long)gid * 32 + l * 8);
        f4 lp01 = locv[0];                         // points 0,1: (x,y,x,y)
        f4 lp23 = locv[1];                         // points 2,3
        f4 aw4  = *(const f4*)(attnw + (long long)gid * 16 + l * 4);

        float lxs[4] = { lp01.x, lp01.z, lp23.x, lp23.z };
        float lys[4] = { lp01.y, lp01.w, lp23.y, lp23.w };
        float aws[4] = { aw4.x,  aw4.y,  aw4.z,  aw4.w  };

        #pragma unroll
        for (int p = 0; p < NUM_POINTS; ++p) {
            float x = lxs[p] * (float)W - 0.5f;
            float y = lys[p] * (float)H - 0.5f;
            float x0f = floorf(x), y0f = floorf(y);
            float wx = x - x0f, wy = y - y0f;
            int x0 = (int)x0f, y0 = (int)y0f;
            float s00 = corner_fetch(vl, y0,     x0,     H, W, lane);
            float s01 = corner_fetch(vl, y0,     x0 + 1, H, W, lane);
            float s10 = corner_fetch(vl, y0 + 1, x0,     H, W, lane);
            float s11 = corner_fetch(vl, y0 + 1, x0 + 1, H, W, lane);
            float sx0 = s00 * (1.f - wx) + s01 * wx;
            float sx1 = s10 * (1.f - wx) + s11 * wx;
            acc += aws[p] * (sx0 * (1.f - wy) + sx1 * wy);
        }
    }
    mid[(long long)bq * EMBED_DIM + h * HEAD_DIM + lane] = acc;
}

// ---------------- launcher ----------------
extern "C" void kernel_launch(void* const* d_in, const int* in_sizes, int n_in,
                              void* d_out, int out_size, void* d_ws, size_t ws_size,
                              hipStream_t stream)
{
    const float* query  = (const float*)d_in[0];
    const float* ref    = (const float*)d_in[1];
    const float* value  = (const float*)d_in[2];
    const float* W_off  = (const float*)d_in[3];
    const float* b_off  = (const float*)d_in[4];
    const float* W_attn = (const float*)d_in[5];
    const float* b_attn = (const float*)d_in[6];
    const float* W_val  = (const float*)d_in[7];
    const float* b_val  = (const float*)d_in[8];
    const float* W_out  = (const float*)d_in[9];
    const float* b_out  = (const float*)d_in[10];
    float* out = (float*)d_out;

    float* ws       = (float*)d_ws;
    float* vproj    = ws;                                                   // 27,226,112
    float* off_raw  = vproj   + (size_t)BS * NUM_HEADS * LEN_V * HEAD_DIM;  //  2,048,000
    float* attn_raw = off_raw + (size_t)BS * LEN_Q * 256;                   //  1,024,000
    float* loc      = attn_raw+ (size_t)BS * LEN_Q * 128;                   //  2,048,000
    float* attnw    = loc     + (size_t)BS * LEN_Q * NUM_HEADS * 16 * 2;    //  1,024,000
    float* mid      = attnw   + (size_t)BS * LEN_Q * NUM_HEADS * 16;        //  2,048,000

    dim3 blk(256);

    // 1) value projection (store head-major for coalesced gathers)
    {
        int M = BS * LEN_V;
        dim3 grid((M + BM - 1) / BM, EMBED_DIM / BN);
        gemm_f16_wmma<1><<<grid, blk, 0, stream>>>(value, W_val, b_val, vproj,
                                                   M, EMBED_DIM, EMBED_DIM);
    }
    // 2) offset projection
    {
        int M = BS * LEN_Q;
        dim3 grid((M + BM - 1) / BM, 256 / BN);
        gemm_f16_wmma<0><<<grid, blk, 0, stream>>>(query, W_off, b_off, off_raw,
                                                   M, 256, EMBED_DIM);
    }
    // 3) attention-logit projection
    {
        int M = BS * LEN_Q;
        dim3 grid((M + BM - 1) / BM, 128 / BN);
        gemm_f16_wmma<0><<<grid, blk, 0, stream>>>(query, W_attn, b_attn, attn_raw,
                                                   M, 128, EMBED_DIM);
    }
    // 4) softmax + sampling locations
    {
        int n = BS * LEN_Q * NUM_HEADS;
        prep_kernel<<<(n + 255) / 256, blk, 0, stream>>>(off_raw, attn_raw, ref,
                                                         loc, attnw);
    }
    // 5) bilinear gather (one wave32 per (b,q,h), lane == channel)
    {
        int nwaves = BS * LEN_Q * NUM_HEADS;
        sample_kernel<<<nwaves / 8, blk, 0, stream>>>(vproj, loc, attnw, mid);
    }
    // 6) output projection -> d_out
    {
        int M = BS * LEN_Q;
        dim3 grid((M + BM - 1) / BM, EMBED_DIM / BN);
        gemm_f16_wmma<0><<<grid, blk, 0, stream>>>(mid, W_out, b_out, out,
                                                   M, EMBED_DIM, EMBED_DIM);
    }
}
